// Ours_34746285425030
// MI455X (gfx1250) — matrix operand order repaired and checked by reading, not compiled, not run on hardware
//
#include <hip/hip_runtime.h>

#define NN 50000
#define NE 800000
#define EMB 256
#define NH 4
#define HE 1024          // NH*EMB
#define NLAY 4
#define NG 1024
#define NTASK 128
#define FCIN 2048
#define TNW 8            // waves per block in gemm_tn

typedef __attribute__((ext_vector_type(16))) __bf16 v16bf;
typedef __attribute__((ext_vector_type(8)))  __bf16 v8bf;
typedef __attribute__((ext_vector_type(8)))  float  v8f;

// ---------------------------------------------------------------- utilities
__device__ __forceinline__ float wave_sum(float v) {
#pragma unroll
  for (int off = 16; off > 0; off >>= 1) v += __shfl_xor(v, off, 32);
  return v;
}

__device__ __forceinline__ v8f wmma_bf16(const v16bf& a, const v16bf& b, const v8f& c) {
  return __builtin_amdgcn_wmma_f32_16x16x32_bf16(false, a, false, b, (short)0, c,
                                                 false, false);
}

// ---------------------------------------------------------------- WMMA GEMM
// C[M,N] = A[M,K](bf16,row-major,lda) * BT[N,K](bf16,row-major,ldb) (+bias) (+C if accum)
// One wave per 16(m) x 64(n) tile: 1 A fragment reused across 4 B fragments.
// K multiple of 32, M multiple of 16, N multiple of 64.
__global__ void gemm_nt(const __bf16* __restrict__ A, int lda,
                        const __bf16* __restrict__ BT, int ldb,
                        float* __restrict__ C, int ldc,
                        int M, int N, int K,
                        const float* __restrict__ bias, int accum) {
  int wid  = (blockIdx.x * blockDim.x + threadIdx.x) >> 5;
  int lane = threadIdx.x & 31;
  int tn = N >> 6;                       // 64-wide column tiles
  int mt = wid / tn, nt = wid - mt * tn;
  if (mt >= (M >> 4)) return;
  long m0 = (long)mt * 16, n0 = (long)nt * 64;
  int  half = lane >> 4;
  int  akb = half * 16;                  // A: elem i -> K = akb+i
  int  bkb = half * 16;                  // B: elem i -> K = bkb+i
  const __bf16* ap = A + (m0 + (lane & 15)) * lda;
  const __bf16* bp = BT + (n0 + (lane & 15)) * ldb;
  const long bstep = (long)16 * ldb;     // next 16-column subtile
  v8f acc0 = {0.f,0.f,0.f,0.f,0.f,0.f,0.f,0.f};
  v8f acc1 = acc0, acc2 = acc0, acc3 = acc0;
  for (int k0 = 0; k0 < K; k0 += 32) {
    v16bf a = *(const v16bf*)(ap + k0 + akb);
    v16bf b0 = *(const v16bf*)(bp + k0 + bkb);
    v16bf b1 = *(const v16bf*)(bp + bstep + k0 + bkb);
    v16bf b2 = *(const v16bf*)(bp + 2 * bstep + k0 + bkb);
    v16bf b3 = *(const v16bf*)(bp + 3 * bstep + k0 + bkb);
    acc0 = wmma_bf16(a, b0, acc0);
    acc1 = wmma_bf16(a, b1, acc1);
    acc2 = wmma_bf16(a, b2, acc2);
    acc3 = wmma_bf16(a, b3, acc3);
  }
  long cnb = n0 + (lane & 15);
#pragma unroll
  for (int j = 0; j < 4; j++) {
    long cn = cnb + j * 16;
    float bv = bias ? bias[cn] : 0.f;
    v8f acc = (j == 0) ? acc0 : (j == 1) ? acc1 : (j == 2) ? acc2 : acc3;
#pragma unroll
    for (int r = 0; r < 8; r++) {
      long cm = m0 + r + half * 8;
      long idx = cm * ldc + cn;
      float v = acc[r] + bv;
      if (accum) v += C[idx];
      C[idx] = v;
    }
  }
}

// C[M,N] += A[K,M](bf16,row-major,lda)^T * B[K,N](bf16,row-major,ldb), split-K atomics.
// LDS-staged: each lane does one contiguous 32B global load per matrix per K-step,
// fragments are then gathered from the wave's private LDS slab (in-order LDS).
__global__ void gemm_tn_atomic(const __bf16* __restrict__ A, int lda,
                               const __bf16* __restrict__ B, int ldb,
                               float* __restrict__ C, int ldc,
                               int M, int N, int K, int kchunk) {
  __shared__ __align__(32) __bf16 sA[TNW][32][16];
  __shared__ __align__(32) __bf16 sB[TNW][32][16];
  int w    = threadIdx.x >> 5;
  int wid  = (blockIdx.x * blockDim.x + threadIdx.x) >> 5;
  int lane = threadIdx.x & 31;
  int tn = N >> 4;
  int mt = wid / tn, nt = wid - mt * tn;
  if (mt >= (M >> 4)) return;
  long m0 = (long)mt * 16, n0 = (long)nt * 16;
  int  half = lane >> 4;
  int  akb = half * 8;
  int  bkb = half * 16;
  int  ml  = lane & 15;
  int kstart = blockIdx.y * kchunk;
  int kend = kstart + kchunk; if (kend > K) kend = K;
  v8f acc = {0.f,0.f,0.f,0.f,0.f,0.f,0.f,0.f};
  for (int k0 = kstart; k0 < kend; k0 += 32) {
    int kr = k0 + lane;                    // this lane stages global row kr
    v16bf av, bv;
#pragma unroll
    for (int i = 0; i < 16; i++) { av[i] = (__bf16)0.f; bv[i] = (__bf16)0.f; }
    if (kr < kend) {
      av = *(const v16bf*)(A + (long)kr * lda + m0);
      bv = *(const v16bf*)(B + (long)kr * ldb + n0);
    }
    *(v16bf*)(&sA[w][lane][0]) = av;
    *(v16bf*)(&sB[w][lane][0]) = bv;
    // LDS instructions from one wave execute in order; only this wave touches sA[w]/sB[w]
    v16bf a, b;
#pragma unroll
    for (int i = 0; i < 8; i++) {
      a[i]     = sA[w][akb + i][ml];
      a[i + 8] = sA[w][16 + akb + i][ml];
    }
#pragma unroll
    for (int i = 0; i < 16; i++) b[i] = sB[w][(i < 8) ? (akb + i) : (16 + akb + i - 8)][ml];
    acc = wmma_bf16(a, b, acc);
  }
  long cn = n0 + ml;
#pragma unroll
  for (int r = 0; r < 8; r++) {
    long cm = m0 + r + half * 8;
    atomicAdd(&C[cm * ldc + cn], acc[r]);
  }
}

// ---------------------------------------------------------------- pointwise
__global__ void atom_encoder_kernel(const int* __restrict__ x,
                                    const float* __restrict__ aemb,
                                    float* __restrict__ h, __bf16* __restrict__ hb) {
  long idx = (long)blockIdx.x * blockDim.x + threadIdx.x;
  if (idx >= (long)NN * EMB) return;
  int n = (int)(idx >> 8), d = (int)(idx & 255);
  float s = 0.f;
#pragma unroll
  for (int f = 0; f < 9; f++) {
    int a = x[n * 9 + f];
    s += aemb[((long)f * 100 + a) * EMB + d];
  }
  h[idx] = s; hb[idx] = (__bf16)s;
}

// out[c*R + r] = (bf16) in[r*Cc + c]
__global__ void transpose_bf16_kernel(const float* __restrict__ in,
                                      __bf16* __restrict__ out, int R, int Cc) {
  long idx = (long)blockIdx.x * blockDim.x + threadIdx.x;
  if (idx >= (long)R * Cc) return;
  int r = (int)(idx / Cc), c = (int)(idx - (long)r * Cc);
  out[(long)c * R + r] = (__bf16)in[idx];
}

__global__ void f32_to_bf16_kernel(const float* __restrict__ in,
                                   __bf16* __restrict__ out, long n) {
  long idx = (long)blockIdx.x * blockDim.x + threadIdx.x;
  if (idx < n) out[idx] = (__bf16)in[idx];
}

// one wave per row of 256: dst = src / ||src||
__global__ void normalize_rows_kernel(const float* __restrict__ src,
                                      __bf16* __restrict__ dst, int rows) {
  int wid = (blockIdx.x * blockDim.x + threadIdx.x) >> 5;
  int lane = threadIdx.x & 31;
  if (wid >= rows) return;
  const float* p = src + (long)wid * EMB;
  float ss = 0.f;
  for (int i = lane; i < EMB; i += 32) { float v = p[i]; ss += v * v; }
  ss = wave_sum(ss);
  float inv = rsqrtf(ss);
  __bf16* q = dst + (long)wid * EMB;
  for (int i = lane; i < EMB; i += 32) q[i] = (__bf16)(p[i] * inv);
}

__global__ void colsum_f32_kernel(const float* __restrict__ A, float* __restrict__ out,
                                  int rows, int cols) {
  int c = blockIdx.x;
  float s = 0.f;
  for (int r = threadIdx.x; r < rows; r += blockDim.x) s += A[(long)r * cols + c];
  __shared__ float red[256];
  red[threadIdx.x] = s; __syncthreads();
  for (int st = 128; st > 0; st >>= 1) {
    if ((int)threadIdx.x < st) red[threadIdx.x] += red[threadIdx.x + st];
    __syncthreads();
  }
  if (threadIdx.x == 0) out[c] = red[0];
}

__global__ void colsum_bf16_kernel(const __bf16* __restrict__ A, float* __restrict__ out,
                                   int rows, int cols) {
  int c = blockIdx.x;
  float s = 0.f;
  for (int r = threadIdx.x; r < rows; r += blockDim.x) s += (float)A[(long)r * cols + c];
  __shared__ float red[256];
  red[threadIdx.x] = s; __syncthreads();
  for (int st = 128; st > 0; st >>= 1) {
    if ((int)threadIdx.x < st) red[threadIdx.x] += red[threadIdx.x + st];
    __syncthreads();
  }
  if (threadIdx.x == 0) out[c] = red[0];
}

// x1b[row] = (num[row] + hsum) / (qn[row].ksum[head] + NN); one wave per (n,head)
__global__ void attn_finalize_kernel(const float* __restrict__ num,
                                     const __bf16* __restrict__ qn,
                                     const float* __restrict__ ksum,
                                     const float* __restrict__ hsum,
                                     __bf16* __restrict__ x1b, int rows) {
  int wid = (blockIdx.x * blockDim.x + threadIdx.x) >> 5;
  int lane = threadIdx.x & 31;
  if (wid >= rows) return;
  int hh = wid & (NH - 1);
  const __bf16* qp = qn + (long)wid * EMB;
  const float* kp = ksum + hh * EMB;
  float d = 0.f;
  for (int i = lane; i < EMB; i += 32) d += (float)qp[i] * kp[i];
  d = wave_sum(d) + (float)NN;
  const float* np = num + (long)wid * EMB;
  __bf16* op = x1b + (long)wid * EMB;
  for (int i = lane; i < EMB; i += 32) op[i] = (__bf16)((np[i] + hsum[i]) / d);
}

// ---------------------------------------------------------------- graph part
__global__ void deg_accum_kernel(const int* __restrict__ row, const int* __restrict__ col,
                                 float* __restrict__ di, float* __restrict__ dj) {
  long e = (long)blockIdx.x * blockDim.x + threadIdx.x;
  if (e >= NE) return;
  atomicAdd(&di[row[e]], 1.f);
  atomicAdd(&dj[col[e]], 1.f);
}

__global__ void deg_inv_kernel(float* __restrict__ deg) {
  long i = (long)blockIdx.x * blockDim.x + threadIdx.x;
  if (i >= NN) return;
  float d = deg[i];
  deg[i] = (d > 0.f) ? rsqrtf(d) : 0.f;
}

__global__ void adj_kernel(const int* __restrict__ row, const int* __restrict__ col,
                           const float* __restrict__ di, const float* __restrict__ dj,
                           float* __restrict__ adj) {
  long e = (long)blockIdx.x * blockDim.x + threadIdx.x;
  if (e >= NE) return;
  adj[e] = di[row[e]] * dj[col[e]];
}

// msg[row[e]] += adj[e] * (bond0[a0]+bond1[a1]+bond2[a2]); block = one edge, 256 threads
__global__ void message_kernel(const int* __restrict__ row, const int* __restrict__ eattr,
                               const float* __restrict__ bond_l,
                               const float* __restrict__ adj, float* __restrict__ msg) {
  int e = blockIdx.x, d = threadIdx.x;
  int a0 = eattr[e * 3 + 0], a1 = eattr[e * 3 + 1], a2 = eattr[e * 3 + 2];
  float v = bond_l[(0 * 10 + a0) * EMB + d] + bond_l[(1 * 10 + a1) * EMB + d] +
            bond_l[(2 * 10 + a2) * EMB + d];
  atomicAdd(&msg[(long)row[e] * EMB + d], adj[e] * v);
}

__global__ void spmm_kernel(const int* __restrict__ row, const int* __restrict__ col,
                            const float* __restrict__ adj,
                            const float* __restrict__ cur, float* __restrict__ out) {
  int e = blockIdx.x, d = threadIdx.x;
  atomicAdd(&out[(long)row[e] * EMB + d], adj[e] * cur[(long)col[e] * EMB + d]);
}

__global__ void add_msg_writeout_kernel(const float* __restrict__ spmm,
                                        const float* __restrict__ msg,
                                        float* __restrict__ outf,
                                        __bf16* __restrict__ x2b, int chunkOff) {
  long idx = (long)blockIdx.x * blockDim.x + threadIdx.x;
  if (idx >= (long)NN * EMB) return;
  int n = (int)(idx >> 8), d = (int)(idx & 255);
  float v = spmm[idx] + msg[idx];
  outf[idx] = v;
  x2b[(long)n * HE + chunkOff + d] = (__bf16)v;
}

__global__ void copy_bf16_chunk_kernel(const __bf16* __restrict__ src,
                                       __bf16* __restrict__ x2b, int chunkOff) {
  long idx = (long)blockIdx.x * blockDim.x + threadIdx.x;
  if (idx >= (long)NN * EMB) return;
  int n = (int)(idx >> 8), d = (int)(idx & 255);
  x2b[(long)n * HE + chunkOff + d] = src[idx];
}

// z+h -> layernorm -> relu -> h (f32) & hb (bf16); one block (256 thr) per node
__global__ void ln_relu_kernel(const float* __restrict__ z, float* __restrict__ h,
                               __bf16* __restrict__ hb,
                               const float* __restrict__ g, const float* __restrict__ b) {
  int n = blockIdx.x, d = threadIdx.x;
  __shared__ float red[256];
  float v = z[(long)n * EMB + d] + h[(long)n * EMB + d];
  red[d] = v; __syncthreads();
  for (int st = 128; st > 0; st >>= 1) {
    if (d < st) red[d] += red[d + st];
    __syncthreads();
  }
  float mu = red[0] * (1.f / EMB);
  __syncthreads();
  float c = v - mu;
  red[d] = c * c; __syncthreads();
  for (int st = 128; st > 0; st >>= 1) {
    if (d < st) red[d] += red[d + st];
    __syncthreads();
  }
  float var = red[0] * (1.f / EMB);
  float o = c * rsqrtf(var + 1e-5f) * g[d] + b[d];
  o = fmaxf(o, 0.f);
  h[(long)n * EMB + d] = o;
  hb[(long)n * EMB + d] = (__bf16)o;
}

__global__ void pool_accum_kernel(const float* __restrict__ h, const int* __restrict__ batch,
                                  float* __restrict__ sums, float* __restrict__ cnts) {
  int n = blockIdx.x, d = threadIdx.x;
  int g = batch[n];
  atomicAdd(&sums[(long)g * EMB + d], h[(long)n * EMB + d]);
  if (d == 0) atomicAdd(&cnts[g], 1.f);
}

__global__ void pool_final_kernel(const float* __restrict__ sums,
                                  const float* __restrict__ cnts,
                                  __bf16* __restrict__ pb) {
  long idx = (long)blockIdx.x * blockDim.x + threadIdx.x;
  if (idx >= (long)NG * EMB) return;
  int g = (int)(idx >> 8);
  pb[idx] = (__bf16)(sums[idx] / fmaxf(cnts[g], 1.f));
}

// ---------------------------------------------------------------- host side
static inline int cdiv(long a, long b) { return (int)((a + b - 1) / b); }

static void launch_gemm_nt(hipStream_t s, const __bf16* A, int lda, const __bf16* BT,
                           int ldb, float* C, int ldc, int M, int N, int K,
                           const float* bias, int accum) {
  long tiles = (long)(M / 16) * (N / 64);   // 16x64 per wave
  gemm_nt<<<cdiv(tiles, 8), 256, 0, s>>>(A, lda, BT, ldb, C, ldc, M, N, K, bias, accum);
}

extern "C" void kernel_launch(void* const* d_in, const int* in_sizes, int n_in,
                              void* d_out, int out_size, void* d_ws, size_t ws_size,
                              hipStream_t stream) {
  (void)in_sizes; (void)n_in; (void)out_size; (void)ws_size;
  const int*   x        = (const int*)d_in[0];
  const int*   eidx     = (const int*)d_in[1];
  const int*   eattr    = (const int*)d_in[2];
  const int*   batch    = (const int*)d_in[3];
  const float* atom_emb = (const float*)d_in[4];
  const float* bond_emb = (const float*)d_in[5];
  const float* Wq_w     = (const float*)d_in[6];
  const float* Wq_b     = (const float*)d_in[7];
  const float* Wk_w     = (const float*)d_in[8];
  const float* Wk_b     = (const float*)d_in[9];
  const float* fc_w     = (const float*)d_in[10];
  const float* fc_b     = (const float*)d_in[11];
  const float* ln_g     = (const float*)d_in[12];
  const float* ln_b     = (const float*)d_in[13];
  const float* out_w    = (const float*)d_in[14];
  const float* out_b    = (const float*)d_in[15];
  const int* row = eidx;
  const int* col = eidx + NE;

  // ---------------- workspace bump allocator
  char* base = (char*)d_ws;
  size_t off = 0;
  auto bump = [&](size_t bytes) -> void* {
    off = (off + 255) & ~(size_t)255;
    void* p = base + off;
    off += bytes;
    return p;
  };
  float*  h     = (float*)bump((size_t)NN * EMB * 4);
  __bf16* hb    = (__bf16*)bump((size_t)NN * EMB * 2);
  float*  tmpf  = (float*)bump((size_t)NN * HE * 4);   // q/k pre-norm, then num
  __bf16* x2b   = (__bf16*)tmpf;                       // alias (used after tmpf dead)
  __bf16* qn    = (__bf16*)bump((size_t)NN * HE * 2);
  __bf16* kn    = (__bf16*)bump((size_t)NN * HE * 2);
  __bf16* x1b   = kn;                                  // alias (used after kn dead)
  float*  curA  = (float*)bump((size_t)NN * EMB * 4);
  float*  curB  = (float*)bump((size_t)NN * EMB * 4);
  float*  msg   = (float*)bump((size_t)NN * EMB * 4);
  float*  zbuf  = (float*)bump((size_t)NN * EMB * 4);
  __bf16* WqT   = (__bf16*)bump((size_t)NLAY * HE * EMB * 2);
  __bf16* WkT   = (__bf16*)bump((size_t)NLAY * HE * EMB * 2);
  __bf16* fcT   = (__bf16*)bump((size_t)NLAY * EMB * FCIN * 2);
  __bf16* outT  = (__bf16*)bump((size_t)NTASK * EMB * 2);
  float*  kvsT  = (float*)bump((size_t)NH * EMB * EMB * 4);
  __bf16* kvsTb = (__bf16*)bump((size_t)NH * EMB * EMB * 2);
  float*  ksum  = (float*)bump((size_t)HE * 4);
  float*  hsum  = (float*)bump((size_t)EMB * 4);
  float*  degi  = (float*)bump((size_t)NN * 4);
  float*  degj  = (float*)bump((size_t)NN * 4);
  float*  adjv  = (float*)bump((size_t)NE * 4);
  float*  psum  = (float*)bump((size_t)NG * EMB * 4);
  float*  cnts  = (float*)bump((size_t)NG * 4);
  __bf16* poolb = (__bf16*)bump((size_t)NG * EMB * 2);

  const long NDE = (long)NN * EMB;            // node-dim elements
  const int  EB  = cdiv(NDE, 256);

  // ---------------- weight prep (bf16, transposed)
  for (int l = 0; l < NLAY; l++) {
    transpose_bf16_kernel<<<cdiv((long)EMB * HE, 256), 256, 0, stream>>>(
        Wq_w + (long)l * EMB * HE, WqT + (long)l * HE * EMB, EMB, HE);
    transpose_bf16_kernel<<<cdiv((long)EMB * HE, 256), 256, 0, stream>>>(
        Wk_w + (long)l * EMB * HE, WkT + (long)l * HE * EMB, EMB, HE);
    transpose_bf16_kernel<<<cdiv((long)FCIN * EMB, 256), 256, 0, stream>>>(
        fc_w + (long)l * FCIN * EMB, fcT + (long)l * EMB * FCIN, FCIN, EMB);
  }
  transpose_bf16_kernel<<<cdiv((long)EMB * NTASK, 256), 256, 0, stream>>>(
      out_w, outT, EMB, NTASK);

  // ---------------- atom encoder
  atom_encoder_kernel<<<EB, 256, 0, stream>>>(x, atom_emb, h, hb);

  // ---------------- degrees / adjacency (layer-invariant)
  hipMemsetAsync(degi, 0, (size_t)NN * 4, stream);
  hipMemsetAsync(degj, 0, (size_t)NN * 4, stream);
  deg_accum_kernel<<<cdiv(NE, 256), 256, 0, stream>>>(row, col, degi, degj);
  deg_inv_kernel<<<cdiv(NN, 256), 256, 0, stream>>>(degi);
  deg_inv_kernel<<<cdiv(NN, 256), 256, 0, stream>>>(degj);
  adj_kernel<<<cdiv(NE, 256), 256, 0, stream>>>(row, col, degi, degj, adjv);

  const int KSPLIT = 32;
  const int kchunk = ((NN + KSPLIT - 1) / KSPLIT + 31) & ~31;

  for (int l = 0; l < NLAY; l++) {
    // ---- attention branch
    colsum_f32_kernel<<<EMB, 256, 0, stream>>>(h, hsum, NN, EMB);
    // q
    launch_gemm_nt(stream, hb, EMB, WqT + (long)l * HE * EMB, EMB, tmpf, HE,
                   NN, HE, EMB, Wq_b + (long)l * HE, 0);
    normalize_rows_kernel<<<cdiv((long)NN * NH, 8), 256, 0, stream>>>(tmpf, qn, NN * NH);
    // k
    launch_gemm_nt(stream, hb, EMB, WkT + (long)l * HE * EMB, EMB, tmpf, HE,
                   NN, HE, EMB, Wk_b + (long)l * HE, 0);
    normalize_rows_kernel<<<cdiv((long)NN * NH, 8), 256, 0, stream>>>(tmpf, kn, NN * NH);
    colsum_bf16_kernel<<<HE, 256, 0, stream>>>(kn, ksum, NN, HE);
    // kvsT[h][d][m] = sum_n hb[n,d] * kn[n,h*EMB+m]   (split-K WMMA with atomics)
    hipMemsetAsync(kvsT, 0, (size_t)NH * EMB * EMB * 4, stream);
    for (int hh = 0; hh < NH; hh++) {
      dim3 grid(cdiv((long)(EMB / 16) * (EMB / 16), TNW), KSPLIT);
      gemm_tn_atomic<<<grid, 32 * TNW, 0, stream>>>(
          hb, EMB, kn + hh * EMB, HE, kvsT + (long)hh * EMB * EMB, EMB,
          EMB, EMB, NN, kchunk);
    }
    f32_to_bf16_kernel<<<cdiv((long)NH * EMB * EMB, 256), 256, 0, stream>>>(
        kvsT, kvsTb, (long)NH * EMB * EMB);
    // num[n,h,d] = sum_m qn[n,h,m] * kvsT[h][d][m]
    for (int hh = 0; hh < NH; hh++) {
      launch_gemm_nt(stream, qn + hh * EMB, HE, kvsTb + (long)hh * EMB * EMB, EMB,
                     tmpf + hh * EMB, HE, NN, EMB, EMB, nullptr, 0);
    }
    attn_finalize_kernel<<<cdiv((long)NN * NH, 8), 256, 0, stream>>>(
        tmpf, qn, ksum, hsum, x1b, NN * NH);   // tmpf dead after this -> x2b alias ok

    // ---- GNN high-order branch
    copy_bf16_chunk_kernel<<<EB, 256, 0, stream>>>(hb, x2b, 0);
    hipMemsetAsync(msg, 0, (size_t)NN * EMB * 4, stream);
    message_kernel<<<NE, 256, 0, stream>>>(row, eattr,
                                           bond_emb + (long)l * 3 * 10 * EMB, adjv, msg);
    hipMemsetAsync(curA, 0, (size_t)NN * EMB * 4, stream);
    spmm_kernel<<<NE, 256, 0, stream>>>(row, col, adjv, h, curA);
    add_msg_writeout_kernel<<<EB, 256, 0, stream>>>(curA, msg, curA, x2b, 1 * EMB);
    hipMemsetAsync(curB, 0, (size_t)NN * EMB * 4, stream);
    spmm_kernel<<<NE, 256, 0, stream>>>(row, col, adjv, curA, curB);
    add_msg_writeout_kernel<<<EB, 256, 0, stream>>>(curB, msg, curB, x2b, 2 * EMB);
    hipMemsetAsync(curA, 0, (size_t)NN * EMB * 4, stream);
    spmm_kernel<<<NE, 256, 0, stream>>>(row, col, adjv, curB, curA);
    add_msg_writeout_kernel<<<EB, 256, 0, stream>>>(curA, msg, curA, x2b, 3 * EMB);

    // ---- FC over concat([x1, x2]) as two accumulating GEMMs
    launch_gemm_nt(stream, x1b, HE, fcT + (long)l * EMB * FCIN, FCIN, zbuf, EMB,
                   NN, EMB, HE, fc_b + (long)l * EMB, 0);
    launch_gemm_nt(stream, x2b, HE, fcT + (long)l * EMB * FCIN + HE, FCIN, zbuf, EMB,
                   NN, EMB, HE, nullptr, 1);

    // ---- residual + layernorm + relu
    ln_relu_kernel<<<NN, 256, 0, stream>>>(zbuf, h, hb,
                                           ln_g + (long)l * EMB, ln_b + (long)l * EMB);
  }

  // ---------------- mean pool + readout
  hipMemsetAsync(psum, 0, (size_t)NG * EMB * 4, stream);
  hipMemsetAsync(cnts, 0, (size_t)NG * 4, stream);
  pool_accum_kernel<<<NN, 256, 0, stream>>>(h, batch, psum, cnts);
  pool_final_kernel<<<cdiv((long)NG * EMB, 256), 256, 0, stream>>>(psum, cnts, poolb);
  launch_gemm_nt(stream, poolb, EMB, outT, EMB, (float*)d_out, NTASK,
                 NG, NTASK, EMB, out_b, 0);
}
